// MultiHeadAttention_8108898255468
// MI455X (gfx1250) — matrix operand order repaired and checked by reading, not compiled, not run on hardware
//
#include <hip/hip_runtime.h>

typedef __bf16 bf16;
typedef __attribute__((ext_vector_type(16))) __bf16 v16bf;
typedef __attribute__((ext_vector_type(8)))  __bf16 v8bf;
typedef __attribute__((ext_vector_type(4)))  __bf16 v4bf;
typedef __attribute__((ext_vector_type(8)))  float  v8f;

#define DIM   1024
#define NH    16
#define HD    64
#define BATCH 2
#define SEQ   2048
#define MTOK  (BATCH*SEQ)   // 4096 tokens

__device__ __forceinline__ v8f vzero8() {
  v8f z;
#pragma unroll
  for (int i = 0; i < 8; ++i) z[i] = 0.0f;
  return z;
}

__device__ __forceinline__ v8f wmma_bf16(v16bf a, v16bf b, v8f c) {
  // D = A(16x32 bf16) * B(32x16 bf16) + C(16x16 f32)
  return __builtin_amdgcn_wmma_f32_16x16x32_bf16(
      /*neg_a=*/false, a, /*neg_b=*/false, b,
      /*c_mod=*/(short)0, c, /*reuse_a=*/false, /*reuse_b=*/false);
}

__device__ __forceinline__ v16bf comb(v8bf lo, v8bf hi) {
  return __builtin_shufflevector(lo, hi, 0,1,2,3,4,5,6,7,8,9,10,11,12,13,14,15);
}

// A fragment at element offset `off` from per-lane base p (row already selected):
// lane half0 holds K {0..7,16..23}, half1 {8..15,24..31}; base includes +half*8.
#define LOAD_A(p, off) comb(*(const v8bf*)((p) + (off)), *(const v8bf*)((p) + (off) + 16))
// B fragment: same K split as A (base includes +half*8)
#define LOAD_B(p, off) comb(*(const v8bf*)((p) + (off)), *(const v8bf*)((p) + (off) + 16))

// one-time Q fragment loader (lane-dependent row)
__device__ __forceinline__ v16bf load_a_frag(const bf16* __restrict__ A, int ld,
                                             int rowBase, int k0) {
  const int lane = threadIdx.x & 31;
  const int half = lane >> 4;
  const bf16* p = A + (size_t)(rowBase + (lane & 15)) * ld + k0 + half * 8;
  return comb(*(const v8bf*)p, *(const v8bf*)(p + 16));
}

// ---------------------------------------------------------------- fp32 -> bf16
__global__ void cvt_bf16_kernel(const float* __restrict__ src,
                                bf16* __restrict__ dst, int n4) {
  int i = blockIdx.x * blockDim.x + threadIdx.x;
  if (i < n4) {
    float4 f = ((const float4*)src)[i];
    v4bf o;
    o[0] = (bf16)f.x; o[1] = (bf16)f.y; o[2] = (bf16)f.z; o[3] = (bf16)f.w;
    ((v4bf*)dst)[i] = o;
  }
}

// ------------------------------------------------------- QKV: [4096,1024]x[3072,1024]^T
// 32x64 tile/wave; a0/a1/b0/b1 double-buffered; b2/b3 loaded in-iteration
// (latency covered by the four b0/b1 WMMAs) to keep register pressure spill-free.
__global__ void qkv_gemm_kernel(const bf16* __restrict__ xb,
                                const bf16* __restrict__ wb,
                                bf16* __restrict__ Qb, bf16* __restrict__ Kb,
                                bf16* __restrict__ Vt) {
  const int waveId = (blockIdx.x * blockDim.x + threadIdx.x) >> 5;
  const int lane = threadIdx.x & 31;
  const int half = lane >> 4, ln = lane & 15;
  const int mTile = waveId & 127;   // 128 tiles of 32 rows (tokens)
  const int nChunk = waveId >> 7;   // 48 chunks of 64 cols
  const int mBase = mTile * 32;
  const int nBase = nChunk * 64;

  const bf16* pA = xb + (size_t)(mBase + ln) * DIM + half * 8;
  const bf16* pB = wb + (size_t)(nBase + ln) * DIM + half * 8;
  const int R = 16 * DIM;           // 16 rows in elements (32768 B)

  v8f acc[2][4];
#pragma unroll
  for (int g = 0; g < 2; ++g)
#pragma unroll
    for (int f = 0; f < 4; ++f) acc[g][f] = vzero8();

  v16bf a0 = LOAD_A(pA, 0);
  v16bf a1 = LOAD_A(pA, R);
  v16bf b0 = LOAD_B(pB, 0);
  v16bf b1 = LOAD_B(pB, R);

  for (int k0 = 32; k0 < DIM; k0 += 32) {
    // current block's late fragments (consumed after 4 WMMAs below)
    v16bf b2 = LOAD_B(pB, 2 * R);
    v16bf b3 = LOAD_B(pB, 3 * R);
    pA += 32; pB += 32;
    // next block's early fragments
    v16bf a0n = LOAD_A(pA, 0);
    v16bf a1n = LOAD_A(pA, R);
    v16bf b0n = LOAD_B(pB, 0);
    v16bf b1n = LOAD_B(pB, R);

    acc[0][0] = wmma_bf16(a0, b0, acc[0][0]);
    acc[1][0] = wmma_bf16(a1, b0, acc[1][0]);
    acc[0][1] = wmma_bf16(a0, b1, acc[0][1]);
    acc[1][1] = wmma_bf16(a1, b1, acc[1][1]);
    acc[0][2] = wmma_bf16(a0, b2, acc[0][2]);
    acc[1][2] = wmma_bf16(a1, b2, acc[1][2]);
    acc[0][3] = wmma_bf16(a0, b3, acc[0][3]);
    acc[1][3] = wmma_bf16(a1, b3, acc[1][3]);

    a0 = a0n; a1 = a1n; b0 = b0n; b1 = b1n;
  }
  // tail block
  {
    v16bf b2 = LOAD_B(pB, 2 * R);
    v16bf b3 = LOAD_B(pB, 3 * R);
    acc[0][0] = wmma_bf16(a0, b0, acc[0][0]);
    acc[1][0] = wmma_bf16(a1, b0, acc[1][0]);
    acc[0][1] = wmma_bf16(a0, b1, acc[0][1]);
    acc[1][1] = wmma_bf16(a1, b1, acc[1][1]);
    acc[0][2] = wmma_bf16(a0, b2, acc[0][2]);
    acc[1][2] = wmma_bf16(a1, b2, acc[1][2]);
    acc[0][3] = wmma_bf16(a0, b3, acc[0][3]);
    acc[1][3] = wmma_bf16(a1, b3, acc[1][3]);
  }

#pragma unroll
  for (int g = 0; g < 2; ++g) {
#pragma unroll
    for (int f = 0; f < 4; ++f) {
      const int n = nBase + f * 16 + ln;         // 0..3071
      const int c = n >> 10;                     // 0=q 1=k 2=v
      const int h = (n & 1023) >> 6;
      const int d = n & 63;
#pragma unroll
      for (int r = 0; r < 8; ++r) {
        const int m = mBase + g * 16 + r + half * 8;   // token index
        const int b = m >> 11, l = m & 2047;
        const size_t bh = (size_t)(b * NH + h);
        const bf16 v = (bf16)acc[g][f][r];
        if (c == 0)      Qb[(bh * SEQ + l) * HD + d] = v;
        else if (c == 1) Kb[(bh * SEQ + l) * HD + d] = v;
        else             Vt[(bh * HD + d) * SEQ + l] = v;
      }
    }
  }
}

// ------------------------------------------------------------- flash attention
// One wave per (bh, 16-query tile); KV streamed in steps of 32 w/ online softmax.
// K+V loads issued together at block top; V latency drains under softmax VALU.
__global__ void attn_kernel(const bf16* __restrict__ Qb, const bf16* __restrict__ Kb,
                            const bf16* __restrict__ Vt, bf16* __restrict__ Ob) {
  __shared__ __align__(16) bf16 ldsP[8][16 * 32];   // per-wave P staging (1KB each)

  const int waveId = (blockIdx.x * blockDim.x + threadIdx.x) >> 5;
  const int wl = (threadIdx.x >> 5) & 7;
  const int lane = threadIdx.x & 31;
  const int half = lane >> 4, ln = lane & 15;
  const int qTile = waveId & 127;   // 128 query tiles of 16
  const int bh = waveId >> 7;       // 0..31

  const bf16* Qp = Qb + ((size_t)bh * SEQ + qTile * 16) * HD;
  // per-lane streaming pointers (advance per kv block); all frag loads use immediates
  const bf16* pK = Kb + (size_t)bh * SEQ * HD + (size_t)ln * HD + half * 8;
  const bf16* pV = Vt + (size_t)bh * HD * SEQ + (size_t)ln * SEQ + half * 8;
  const int KROW16 = 16 * HD;       // 1024 elements
  const int VCOL16 = 16 * SEQ;      // 32768 elements

  const v16bf qf0 = load_a_frag(Qp, HD, 0, 0);
  const v16bf qf1 = load_a_frag(Qp, HD, 0, 32);

  float m_i[8], l_i[8];
  v8f acc[4];
#pragma unroll
  for (int r = 0; r < 8; ++r) { m_i[r] = -1e30f; l_i[r] = 0.0f; }
#pragma unroll
  for (int f = 0; f < 4; ++f) acc[f] = vzero8();

  const float scale = 0.125f;  // 1/sqrt(64)

  for (int kv0 = 0; kv0 < SEQ; kv0 += 32) {
    // ---- issue all K and V loads for this block (K first; V drains under softmax)
    v16bf kb0 = LOAD_B(pK, 0);
    v16bf kb1 = LOAD_B(pK, 32);
    v16bf kb2 = LOAD_B(pK, KROW16);
    v16bf kb3 = LOAD_B(pK, KROW16 + 32);
    v16bf vb0 = LOAD_B(pV, 0);
    v16bf vb1 = LOAD_B(pV, VCOL16);
    v16bf vb2 = LOAD_B(pV, 2 * VCOL16);
    v16bf vb3 = LOAD_B(pV, 3 * VCOL16);
    // prefetch next block (emits global_prefetch_b8; offsets fold to immediates)
    __builtin_prefetch(pK + 32 * HD, 0, 0);
    __builtin_prefetch(pV + 32, 0, 0);
    pK += 32 * HD;
    pV += 32;

    // ---- scores S = Q K^T for 16q x 32kv
    v8f s0 = vzero8(), s1 = vzero8();
    s0 = wmma_bf16(qf0, kb0, s0);
    s0 = wmma_bf16(qf1, kb1, s0);
    s1 = wmma_bf16(qf0, kb2, s1);
    s1 = wmma_bf16(qf1, kb3, s1);

    // ---- online softmax: row max across the 16 lanes of each half
    float mnew[8], alpha[8];
#pragma unroll
    for (int r = 0; r < 8; ++r) {
      const float a = s0[r] * scale, b = s1[r] * scale;
      s0[r] = a; s1[r] = b;
      float v = fmaxf(a, b);
      v = fmaxf(v, __shfl_xor(v, 1));
      v = fmaxf(v, __shfl_xor(v, 2));
      v = fmaxf(v, __shfl_xor(v, 4));
      v = fmaxf(v, __shfl_xor(v, 8));
      mnew[r] = fmaxf(m_i[r], v);
      alpha[r] = __expf(m_i[r] - mnew[r]);
      m_i[r] = mnew[r];
    }
#pragma unroll
    for (int r = 0; r < 8; ++r) {
      const float p0 = __expf(s0[r] - mnew[r]);
      const float p1 = __expf(s1[r] - mnew[r]);
      s0[r] = p0; s1[r] = p1;
      float v = p0 + p1;
      v += __shfl_xor(v, 1);
      v += __shfl_xor(v, 2);
      v += __shfl_xor(v, 4);
      v += __shfl_xor(v, 8);
      l_i[r] = l_i[r] * alpha[r] + v;
    }
#pragma unroll
    for (int f = 0; f < 4; ++f)
#pragma unroll
      for (int r = 0; r < 8; ++r) acc[f][r] *= alpha[r];

    // ---- transpose P (C-layout -> A-layout) via per-wave LDS (same-wave DS is in-order)
    bf16* pl = ldsP[wl];
#pragma unroll
    for (int r = 0; r < 8; ++r) {
      pl[(r + 8 * half) * 32 + ln]      = (bf16)s0[r];
      pl[(r + 8 * half) * 32 + 16 + ln] = (bf16)s1[r];
    }
    v8bf lo = *(const v8bf*)(pl + ln * 32 + half * 8);
    v8bf hi = *(const v8bf*)(pl + ln * 32 + half * 8 + 16);
    v16bf pf = comb(lo, hi);

    // ---- O += P V   (V pre-transposed: columns of V contiguous in kv)
    acc[0] = wmma_bf16(pf, vb0, acc[0]);
    acc[1] = wmma_bf16(pf, vb1, acc[1]);
    acc[2] = wmma_bf16(pf, vb2, acc[2]);
    acc[3] = wmma_bf16(pf, vb3, acc[3]);
  }

  // ---- normalize, write [B, L, H*D] bf16
  const int b = bh >> 4, h = bh & 15;
#pragma unroll
  for (int f = 0; f < 4; ++f) {
    const int d = f * 16 + ln;
#pragma unroll
    for (int r = 0; r < 8; ++r) {
      const int l = qTile * 16 + r + 8 * half;
      const float v = acc[f][r] / l_i[r];
      Ob[((size_t)(b * SEQ + l)) * DIM + h * HD + d] = (bf16)v;
    }
  }
}

// --------------------------------------------------- out = attn @ Wp^T + bias (fp32)
__global__ void proj_gemm_kernel(const bf16* __restrict__ Ab,
                                 const bf16* __restrict__ Wb,
                                 const float* __restrict__ bias,
                                 float* __restrict__ out) {
  const int waveId = (blockIdx.x * blockDim.x + threadIdx.x) >> 5;
  const int lane = threadIdx.x & 31;
  const int half = lane >> 4, ln = lane & 15;
  const int mTile = waveId & 127;   // 128 tiles of 32 rows
  const int nChunk = waveId >> 7;   // 16 chunks of 64
  const int mBase = mTile * 32;
  const int nBase = nChunk * 64;

  const bf16* pA = Ab + (size_t)(mBase + ln) * DIM + half * 8;
  const bf16* pB = Wb + (size_t)(nBase + ln) * DIM + half * 8;
  const int R = 16 * DIM;

  v8f acc[2][4];
#pragma unroll
  for (int g = 0; g < 2; ++g)
#pragma unroll
    for (int f = 0; f < 4; ++f) acc[g][f] = vzero8();

  v16bf a0 = LOAD_A(pA, 0);
  v16bf a1 = LOAD_A(pA, R);
  v16bf b0 = LOAD_B(pB, 0);
  v16bf b1 = LOAD_B(pB, R);

  for (int k0 = 32; k0 < DIM; k0 += 32) {
    v16bf b2 = LOAD_B(pB, 2 * R);
    v16bf b3 = LOAD_B(pB, 3 * R);
    pA += 32; pB += 32;
    v16bf a0n = LOAD_A(pA, 0);
    v16bf a1n = LOAD_A(pA, R);
    v16bf b0n = LOAD_B(pB, 0);
    v16bf b1n = LOAD_B(pB, R);

    acc[0][0] = wmma_bf16(a0, b0, acc[0][0]);
    acc[1][0] = wmma_bf16(a1, b0, acc[1][0]);
    acc[0][1] = wmma_bf16(a0, b1, acc[0][1]);
    acc[1][1] = wmma_bf16(a1, b1, acc[1][1]);
    acc[0][2] = wmma_bf16(a0, b2, acc[0][2]);
    acc[1][2] = wmma_bf16(a1, b2, acc[1][2]);
    acc[0][3] = wmma_bf16(a0, b3, acc[0][3]);
    acc[1][3] = wmma_bf16(a1, b3, acc[1][3]);

    a0 = a0n; a1 = a1n; b0 = b0n; b1 = b1n;
  }
  {
    v16bf b2 = LOAD_B(pB, 2 * R);
    v16bf b3 = LOAD_B(pB, 3 * R);
    acc[0][0] = wmma_bf16(a0, b0, acc[0][0]);
    acc[1][0] = wmma_bf16(a1, b0, acc[1][0]);
    acc[0][1] = wmma_bf16(a0, b1, acc[0][1]);
    acc[1][1] = wmma_bf16(a1, b1, acc[1][1]);
    acc[0][2] = wmma_bf16(a0, b2, acc[0][2]);
    acc[1][2] = wmma_bf16(a1, b2, acc[1][2]);
    acc[0][3] = wmma_bf16(a0, b3, acc[0][3]);
    acc[1][3] = wmma_bf16(a1, b3, acc[1][3]);
  }

#pragma unroll
  for (int g = 0; g < 2; ++g) {
#pragma unroll
    for (int f = 0; f < 4; ++f) {
      const int n = nBase + f * 16 + ln;
      const float bn = bias[n];
#pragma unroll
      for (int r = 0; r < 8; ++r) {
        const int m = mBase + g * 16 + r + half * 8;
        out[(size_t)m * DIM + n] = acc[g][f][r] + bn;
      }
    }
  }
}

extern "C" void kernel_launch(void* const* d_in, const int* in_sizes, int n_in,
                              void* d_out, int out_size, void* d_ws, size_t ws_size,
                              hipStream_t stream) {
  const float* x      = (const float*)d_in[0];  // [2,2048,1024]
  const float* w_qkv  = (const float*)d_in[1];  // [3072,1024]
  const float* w_proj = (const float*)d_in[2];  // [1024,1024]
  const float* b_proj = (const float*)d_in[3];  // [1024]
  float* out = (float*)d_out;                   // [2,2048,1024]

  char* ws = (char*)d_ws;
  size_t o = 0;
  bf16* xb     = (bf16*)(ws + o); o += (size_t)MTOK * DIM * 2;        // 8 MB
  bf16* wqkvb  = (bf16*)(ws + o); o += (size_t)3 * DIM * DIM * 2;     // 6 MB
  bf16* wprojb = (bf16*)(ws + o); o += (size_t)DIM * DIM * 2;         // 2 MB
  bf16* Qb     = (bf16*)(ws + o); o += (size_t)BATCH * NH * SEQ * HD * 2;
  bf16* Kb     = (bf16*)(ws + o); o += (size_t)BATCH * NH * SEQ * HD * 2;
  bf16* Vt     = (bf16*)(ws + o); o += (size_t)BATCH * NH * HD * SEQ * 2;
  bf16* Ob     = (bf16*)(ws + o); o += (size_t)MTOK * DIM * 2;

  {
    int n4 = MTOK * DIM / 4;
    cvt_bf16_kernel<<<(n4 + 255) / 256, 256, 0, stream>>>(x, xb, n4);
  }
  {
    int n4 = 3 * DIM * DIM / 4;
    cvt_bf16_kernel<<<(n4 + 255) / 256, 256, 0, stream>>>(w_qkv, wqkvb, n4);
  }
  {
    int n4 = DIM * DIM / 4;
    cvt_bf16_kernel<<<(n4 + 255) / 256, 256, 0, stream>>>(w_proj, wprojb, n4);
  }

  // 128 mTiles * 48 nChunks = 6144 waves, 8 waves/block
  qkv_gemm_kernel<<<768, 256, 0, stream>>>(xb, wqkvb, Qb, Kb, Vt);
  // 32 bh * 128 qTiles = 4096 waves
  attn_kernel<<<512, 256, 0, stream>>>(Qb, Kb, Vt, Ob);
  // 128 mTiles * 16 nChunks = 2048 waves
  proj_gemm_kernel<<<256, 256, 0, stream>>>(Ob, wprojb, b_proj, out);
}
